// TextEncoder_62328565399969
// MI455X (gfx1250) — matrix operand order repaired and checked by reading, not compiled, not run on hardware
//
#include <hip/hip_runtime.h>
#include <math.h>

#define B_     4
#define T_     2048
#define H_     768
#define HEADS_ 4
#define D_     768
#define N_     (HEADS_ * D_)   /* 3072 */
#define WIN_   2
#define K_     5
#define EPS_   1e-5f
#define SLOPE_ 0.2f

typedef __attribute__((ext_vector_type(16))) _Float16 v16h;
typedef __attribute__((ext_vector_type(8)))  float    v8f;
typedef __attribute__((ext_vector_type(4)))  int      v4i;

union Frag { v16h v; uint4 q[2]; };
union H4   { _Float16 h[4]; uint2 u2; };

typedef __attribute__((address_space(1))) v4i* gas_v4i;   // global (AS1)
typedef __attribute__((address_space(3))) v4i* las_v4i;   // LDS    (AS3)

// ---- CDNA5 async global->LDS copy (16B per lane), ASYNCcnt-tracked ---------
__device__ __forceinline__ void async_copy16(const _Float16* g, _Float16* l) {
#if __has_builtin(__builtin_amdgcn_global_load_async_to_lds_b128)
  __builtin_amdgcn_global_load_async_to_lds_b128((gas_v4i)g, (las_v4i)l, 0, 0);
#else
  const unsigned           lo = (unsigned)(uintptr_t)l;        // LDS byte offset
  const unsigned long long ga = (unsigned long long)(uintptr_t)g;
  asm volatile("global_load_async_to_lds_b128 %0, %1, off" :: "v"(lo), "v"(ga) : "memory");
#endif
}
__device__ __forceinline__ void wait_async_le4() {
#if __has_builtin(__builtin_amdgcn_s_wait_asynccnt)
  __builtin_amdgcn_s_wait_asynccnt(4);
#else
  asm volatile("s_wait_asynccnt 0x4" ::: "memory");
#endif
}
__device__ __forceinline__ void wait_async_0() {
#if __has_builtin(__builtin_amdgcn_s_wait_asynccnt)
  __builtin_amdgcn_s_wait_asynccnt(0);
#else
  asm volatile("s_wait_asynccnt 0x0" ::: "memory");
#endif
}

// ---------------------------------------------------------------------------
// One-time per-layer precision/layout prep:
//   Xh[m][k]  = (f16) X[m][k]           (m = B*T, k = H)
//   Wt[n][k]  = (f16) W[k][n]           (transposed so B-tiles are k-contig)
// ---------------------------------------------------------------------------
__global__ void cvt_x_f16(const float* __restrict__ X, _Float16* __restrict__ Xh) {
  const int i = (blockIdx.x * 256 + threadIdx.x) * 4;
  const float4 f = *reinterpret_cast<const float4*>(X + i);
  H4 p;
  p.h[0] = (_Float16)f.x; p.h[1] = (_Float16)f.y;
  p.h[2] = (_Float16)f.z; p.h[3] = (_Float16)f.w;
  *reinterpret_cast<uint2*>(Xh + i) = p.u2;
}

__global__ void cvt_w_f16t(const float* __restrict__ Wl, _Float16* __restrict__ Wt) {
  __shared__ float t[32][33];
  const int tx = threadIdx.x, ty = threadIdx.y;          // block (32,8)
  const int n0 = blockIdx.x * 32, k0 = blockIdx.y * 32;
  for (int i = 0; i < 4; ++i)
    t[ty + i * 8][tx] = Wl[(size_t)(k0 + ty + i * 8) * N_ + n0 + tx];
  __syncthreads();
  for (int i = 0; i < 4; ++i)
    Wt[(size_t)(n0 + ty + i * 8) * H_ + k0 + tx] = (_Float16)t[tx][ty + i * 8];
}

// ---------------------------------------------------------------------------
// GEMM: h[bt, n] = sum_k Xh[bt, k] * Wt[n, k],  M=8192, K=768, N=3072
// 128x128 block tile, BK=32, f16 operands async-copied to LDS, double-buffered.
// 8 waves: wave (wm, wn) owns rows wm*64..+63, cols wn*32..+31 -> 4x2 WMMA tiles.
// ---------------------------------------------------------------------------
__launch_bounds__(256, 2)
__global__ void gat_gemm(const _Float16* __restrict__ Xh,
                         const _Float16* __restrict__ Wt,
                         float* __restrict__ Hout) {
  __shared__ __align__(16) _Float16 lA[2][128][40];   // [m][k], 8-half pad
  __shared__ __align__(16) _Float16 lB[2][128][40];   // [n][k], 8-half pad

  const int tid  = threadIdx.x;
  const int lane = tid & 31;
  const int wid  = tid >> 5;
  const int wm   = wid & 1;
  const int wn   = wid >> 1;
  const int m0   = blockIdx.y * 128;
  const int n0   = blockIdx.x * 128;
  const int lhalf = (lane < 16) ? 0 : 1;

  // copy role: thread -> (row = tid/4 and +64, 16B quarter q = tid%4)
  const int rC = tid >> 2, qC = tid & 3;
  const _Float16* gA = Xh + (size_t)(m0 + rC) * H_ + qC * 8;
  const _Float16* gB = Wt + (size_t)(n0 + rC) * H_ + qC * 8;

  v8f acc[4][2];
  for (int i = 0; i < 4; ++i)
    for (int j = 0; j < 2; ++j)
      for (int e = 0; e < 8; ++e) acc[i][j][e] = 0.0f;

  auto issue = [&](int ks, int bu) {
    const int ko = ks * 32;
    async_copy16(gA + ko,                       &lA[bu][rC][qC * 8]);
    async_copy16(gA + ko + (size_t)64 * H_,     &lA[bu][rC + 64][qC * 8]);
    async_copy16(gB + ko,                       &lB[bu][rC][qC * 8]);
    async_copy16(gB + ko + (size_t)64 * H_,     &lB[bu][rC + 64][qC * 8]);
  };

  issue(0, 0);
  for (int ks = 0; ks < H_ / 32; ++ks) {
    const int bu = ks & 1;
    if (ks + 1 < H_ / 32) { issue(ks + 1, bu ^ 1); wait_async_le4(); }
    else                  { wait_async_0(); }
    __syncthreads();

    // B frags: lane n<16 -> col n, K0..15 ; lane n>=16 -> col n-16, K16..31
    Frag bfr[2];
    for (int nt = 0; nt < 2; ++nt) {
      const int col  = wn * 32 + nt * 16 + (lane & 15);
      const int koff = lhalf ? 16 : 0;
      bfr[nt].q[0] = *reinterpret_cast<const uint4*>(&lB[bu][col][koff]);
      bfr[nt].q[1] = *reinterpret_cast<const uint4*>(&lB[bu][col][koff + 8]);
    }
    // A frags: lane m<16 -> K0..15 ; m>=16 -> K16..31
    for (int mt = 0; mt < 4; ++mt) {
      Frag afr;
      const int row  = wm * 64 + mt * 16 + (lane & 15);
      const int koff = lhalf ? 16 : 0;
      afr.q[0] = *reinterpret_cast<const uint4*>(&lA[bu][row][koff]);
      afr.q[1] = *reinterpret_cast<const uint4*>(&lA[bu][row][koff + 8]);
      for (int nt = 0; nt < 2; ++nt) {
        acc[mt][nt] = __builtin_amdgcn_wmma_f32_16x16x32_f16(
            false, afr.v, false, bfr[nt].v,
            (short)0, acc[mt][nt], false, false);
      }
    }
    __syncthreads();
  }

  // C layout: VGPR r -> M = base + r (+8 for upper lane half), N = lane&15
  for (int mt = 0; mt < 4; ++mt)
    for (int nt = 0; nt < 2; ++nt) {
      const int rb  = m0 + wm * 64 + mt * 16 + (lhalf ? 8 : 0);
      const int col = n0 + wn * 32 + nt * 16 + (lane & 15);
      for (int r = 0; r < 8; ++r)
        Hout[(size_t)(rb + r) * N_ + col] = acc[mt][nt][r];
    }
}

// ---------------------------------------------------------------------------
// asrc[bt,head] = h[bt,head,:] . a_src[head,:]  (same for a_dst); 1 wave each
// ---------------------------------------------------------------------------
__global__ void gat_rowdot(const float* __restrict__ Hb,
                           const float* __restrict__ Asrc,
                           const float* __restrict__ Adst,
                           float* __restrict__ sOut,
                           float* __restrict__ dOut) {
  const int lane = threadIdx.x & 31;
  const int gw   = blockIdx.x * 8 + (threadIdx.x >> 5);  // 0 .. B*T*HEADS-1
  const int head = gw & (HEADS_ - 1);
  const float* hr = Hb + (size_t)gw * D_;
  const float* as = Asrc + head * D_;
  const float* ad = Adst + head * D_;
  float ss = 0.f, dd = 0.f;
  for (int j = 0; j < D_ / 32; ++j) {
    const float v = hr[lane + j * 32];
    ss += v * as[lane + j * 32];
    dd += v * ad[lane + j * 32];
  }
  for (int off = 16; off > 0; off >>= 1) {
    ss += __shfl_xor(ss, off);
    dd += __shfl_xor(dd, off);
  }
  if (lane == 0) { sOut[gw] = ss; dOut[gw] = dd; }
}

// ---------------------------------------------------------------------------
// windowed attention + head-mean + bias + residual + layernorm; 1 block/token
// ---------------------------------------------------------------------------
__launch_bounds__(256)
__global__ void gat_attn_ln(const float* __restrict__ Hb,
                            const float* __restrict__ sV,
                            const float* __restrict__ dV,
                            const float* __restrict__ bias,
                            const float* __restrict__ Xin,
                            const float* __restrict__ gamma,
                            const float* __restrict__ beta,
                            float* __restrict__ Xout) {
  __shared__ float s_at[HEADS_][K_];
  __shared__ float s_sum[8], s_sq[8];
  __shared__ float s_mr[2];
  const int tid = threadIdx.x;
  const int bt  = blockIdx.x;
  const int t   = bt & (T_ - 1);

  if (tid < HEADS_ * K_) {
    const int head = tid / K_, k = tid % K_;
    const int tt = t + k - WIN_;
    float sc = -1e30f;
    if (tt >= 0 && tt < T_) {
      const float e = sV[(bt + (k - WIN_)) * HEADS_ + head] + dV[bt * HEADS_ + head];
      sc = (e > 0.f) ? e : SLOPE_ * e;
    }
    s_at[head][k] = sc;
  }
  __syncthreads();
  if (tid < HEADS_) {
    float mx = -1e30f;
    for (int k = 0; k < K_; ++k) mx = fmaxf(mx, s_at[tid][k]);
    float w[K_], sum = 0.f;
    for (int k = 0; k < K_; ++k) {
      float e = __expf(s_at[tid][k] - mx);
      if (s_at[tid][k] <= -1e29f) e = 0.f;   // masked
      w[k] = e; sum += e;
    }
    const float inv = 1.f / sum;
    for (int k = 0; k < K_; ++k) s_at[tid][k] = w[k] * inv;
  }
  __syncthreads();

  float z[3], lsum = 0.f, lsq = 0.f;
  for (int j = 0; j < 3; ++j) {
    const int d = tid + j * 256;
    float acc = 0.f;
    for (int head = 0; head < HEADS_; ++head)
      for (int k = 0; k < K_; ++k) {
        int tt = t + k - WIN_;
        tt = tt < 0 ? 0 : (tt >= T_ ? T_ - 1 : tt);     // masked weight is 0
        const int btn = bt + (tt - t);
        acc += s_at[head][k] * Hb[((size_t)btn * HEADS_ + head) * D_ + d];
      }
    const float y  = acc * (1.f / HEADS_) + bias[d];
    const float zz = y + Xin[(size_t)bt * H_ + d];
    z[j] = zz; lsum += zz; lsq += zz * zz;
  }
  for (int off = 16; off > 0; off >>= 1) {
    lsum += __shfl_xor(lsum, off);
    lsq  += __shfl_xor(lsq,  off);
  }
  if ((tid & 31) == 0) { s_sum[tid >> 5] = lsum; s_sq[tid >> 5] = lsq; }
  __syncthreads();
  if (tid == 0) {
    float S = 0.f, Q = 0.f;
    for (int i = 0; i < 8; ++i) { S += s_sum[i]; Q += s_sq[i]; }
    const float m   = S / (float)H_;
    const float var = Q / (float)H_ - m * m;
    s_mr[0] = m; s_mr[1] = rsqrtf(var + EPS_);
  }
  __syncthreads();
  const float m = s_mr[0], r = s_mr[1];
  for (int j = 0; j < 3; ++j) {
    const int d = tid + j * 256;
    Xout[(size_t)bt * H_ + d] = (z[j] - m) * r * gamma[d] + beta[d];
  }
}

// ---------------------------------------------------------------------------
extern "C" void kernel_launch(void* const* d_in, const int* in_sizes, int n_in,
                              void* d_out, int out_size, void* d_ws, size_t ws_size,
                              hipStream_t stream) {
  const float* x     = (const float*)d_in[0];
  const float* W     = (const float*)d_in[1];
  const float* a_src = (const float*)d_in[2];
  const float* a_dst = (const float*)d_in[3];
  const float* bias  = (const float*)d_in[4];
  const float* gamma = (const float*)d_in[5];
  const float* beta  = (const float*)d_in[6];
  float* out = (float*)d_out;

  char* ws = (char*)d_ws;
  float* xbuf = (float*)ws;       ws += (size_t)B_ * T_ * H_ * sizeof(float);
  float* hbuf = (float*)ws;       ws += (size_t)B_ * T_ * HEADS_ * D_ * sizeof(float);
  float* sbuf = (float*)ws;       ws += (size_t)B_ * T_ * HEADS_ * sizeof(float);
  float* dbuf = (float*)ws;       ws += (size_t)B_ * T_ * HEADS_ * sizeof(float);
  _Float16* Xh = (_Float16*)ws;   ws += (size_t)B_ * T_ * H_ * sizeof(_Float16);
  _Float16* Wt = (_Float16*)ws;

  const dim3 ggrid(N_ / 128, (B_ * T_) / 128);   // (24, 64)
  for (int l = 0; l < 3; ++l) {
    const float* xin = (l == 0) ? x : xbuf;
    float* xout = (l == 2) ? out : xbuf;         // in-place safe: block touches only its own token
    cvt_x_f16<<<(B_ * T_ * H_) / 1024, 256, 0, stream>>>(xin, Xh);
    cvt_w_f16t<<<dim3(N_ / 32, H_ / 32), dim3(32, 8), 0, stream>>>(
        W + (size_t)l * H_ * N_, Wt);
    gat_gemm<<<ggrid, 256, 0, stream>>>(Xh, Wt, hbuf);
    gat_rowdot<<<(B_ * T_ * HEADS_) / 8, 256, 0, stream>>>(
        hbuf, a_src + (size_t)l * HEADS_ * D_, a_dst + (size_t)l * HEADS_ * D_,
        sbuf, dbuf);
    gat_attn_ln<<<B_ * T_, 256, 0, stream>>>(
        hbuf, sbuf, dbuf, bias + (size_t)l * D_, xin,
        gamma + (size_t)l * H_, beta + (size_t)l * H_, xout);
  }
}
